// HieraMaskUnitAttention_17961553232052
// MI455X (gfx1250) — matrix operand order repaired and checked by reading, not compiled, not run on hardware
//
#include <hip/hip_runtime.h>
#include <hip/hip_bf16.h>

// ---------------------------------------------------------------------------
// Hiera mask-unit attention for MI455X (gfx1250, wave32, WMMA).
// Pipeline: [QKV gemm bf16] -> [windowed attention, WMMA + softmax] -> [proj gemm]
// Matrix math on v_wmma_f32_16x16x32_bf16 (f32 accum); bf16 staging via
// global_load_async_to_lds_b128 (ASYNCcnt) where no conversion is needed.
// ---------------------------------------------------------------------------

typedef __bf16 bf16;
typedef __attribute__((ext_vector_type(16))) __bf16 v16bf;
typedef __attribute__((ext_vector_type(8)))  float  v8f;

static_assert(sizeof(bf16) == 2, "bf16 size");

constexpr int Bb  = 8;
constexpr int Tt  = 8192;
constexpr int Dd  = 768;
constexpr int Hh  = 12;
constexpr int HD  = 64;
constexpr int NW  = 32;     // windows
constexpr int SEQ = 256;    // tokens per window (pre-pool)
constexpr int QL  = 64;     // pooled query length
constexpr int QKV_N = 3 * Dd;        // 2304
constexpr float ATT_SCALE = 0.125f;  // 64^-0.5

// ---- CDNA5 async global->LDS copy (per-lane 16B; INST_OFFSET adds to both
// the LDS and the global address, CDNA5 ISA) ----
template <int OFF>
__device__ __forceinline__ void async_ld_b128(unsigned lds_off, const void* g) {
  asm volatile("global_load_async_to_lds_b128 %0, %1, off offset:%2"
               :: "v"(lds_off), "v"(g), "i"(OFF)
               : "memory");
}
__device__ __forceinline__ void wait_async0() {
  asm volatile("s_wait_asynccnt 0" ::: "memory");
}
__device__ __forceinline__ unsigned lds_off32(const void* p) {
  // generic LDS pointer: addr[31:0] is the LDS byte offset
  return (unsigned)(unsigned long long)p;
}

// ---- WMMA fragment gathers (wave32; k = 16*(lane/16) + element in A and B) ----

// A 16x32 bf16: lanes 0-15 => M=lane, K=0..15; lanes 16-31 => K=16..31
__device__ __forceinline__ v16bf frag_a(const bf16* p0, int ld) {
  const int lane = threadIdx.x & 31;
  const bf16* p = p0 + (lane & 15) * ld + ((lane & 16) ? 16 : 0);
  v16bf a;
#pragma unroll
  for (int e = 0; e < 8; ++e) a[e] = p[e];
#pragma unroll
  for (int e = 0; e < 8; ++e) a[e + 8] = p[e + 8];
  return a;
}

// B 32x16 bf16 from [n][k] storage (NT operand: rows of K / weight rows)
__device__ __forceinline__ v16bf frag_b_nt(const bf16* p0, int ld) {
  const int lane = threadIdx.x & 31;
  const bf16* p = p0 + (lane & 15) * ld + ((lane & 16) ? 16 : 0);
  v16bf b;
#pragma unroll
  for (int e = 0; e < 16; ++e) b[e] = p[e];
  return b;
}

// B 32x16 bf16 from [k][n] storage (NN operand: V)
__device__ __forceinline__ v16bf frag_b_nn(const bf16* p0, int ld) {
  const int lane = threadIdx.x & 31;
  const bf16* p = p0 + (lane & 15) + ((lane & 16) ? 16 * ld : 0);
  v16bf b;
#pragma unroll
  for (int e = 0; e < 16; ++e) b[e] = p[e * ld];
  return b;
}

__device__ __forceinline__ v8f wmma_bf16(v16bf a, v16bf b, v8f c) {
  return __builtin_amdgcn_wmma_f32_16x16x32_bf16(false, a, false, b, (short)0, c,
                                                 false, false);
}

// ---------------------------------------------------------------------------
// GEMM: C[M,N] = A[M,K] * B[N,K]^T + bias[N]   (both operands row-contiguous in K)
// Tile 128x64, K-chunk 32. 256 threads = 8 waves; wave w -> rows [16w,16w+16).
// bf16-A path stages A via async global->LDS; f32 paths convert through VGPRs.
// ---------------------------------------------------------------------------
template <bool A_F32, bool OUT_BF16>
__global__ __launch_bounds__(256) void gemm_nt_bias(
    const void* __restrict__ Ap, const float* __restrict__ Bw,
    const float* __restrict__ bias, void* __restrict__ Op,
    int M, int N, int K) {
  constexpr int TM = 128, TN = 64, TK = 32, LD = 40;  // LD pad: 80B rows (16B aligned)
  __shared__ __align__(16) bf16 As[TM * LD];
  __shared__ __align__(16) bf16 Bs[TN * LD];

  const int tid  = threadIdx.x;
  const int wv   = tid >> 5;
  const int lane = tid & 31;
  const int m0   = blockIdx.x * TM;
  const int n0   = blockIdx.y * TN;

  v8f acc[4] = {};

  for (int k0 = 0; k0 < K; k0 += TK) {
    // stage A chunk (128x32) -> bf16 LDS
    {
      const int row = tid >> 1, hf = tid & 1;
      bf16* dst = &As[row * LD + hf * 16];
      if constexpr (A_F32) {
        const float* src = (const float*)Ap + (size_t)(m0 + row) * K + k0 + hf * 16;
        if (k0 + TK < K) __builtin_prefetch(src + TK, 0, 0);
        bf16 tmp[16];
#pragma unroll
        for (int j = 0; j < 16; ++j) tmp[j] = (bf16)src[j];
        ((uint4*)dst)[0] = ((const uint4*)tmp)[0];
        ((uint4*)dst)[1] = ((const uint4*)tmp)[1];
      } else {
        const bf16* src = (const bf16*)Ap + (size_t)(m0 + row) * K + k0 + hf * 16;
        const unsigned lo = lds_off32(dst);
        async_ld_b128<0>(lo, src);
        async_ld_b128<16>(lo, src);
        if (k0 + TK < K) __builtin_prefetch(src + TK, 0, 0);
      }
    }
    // stage B chunk (64x32) f32 weights -> bf16 LDS
    {
      const int row = tid >> 2, q = tid & 3;
      const float* src = Bw + (size_t)(n0 + row) * K + k0 + q * 8;
      if (k0 + TK < K) __builtin_prefetch(src + TK, 0, 0);
      bf16 tmp[8];
#pragma unroll
      for (int j = 0; j < 8; ++j) tmp[j] = (bf16)src[j];
      *((uint4*)&Bs[row * LD + q * 8]) = *((const uint4*)tmp);
    }
    if constexpr (!A_F32) wait_async0();
    __syncthreads();

    v16bf a = frag_a(&As[(wv * 16) * LD], LD);
#pragma unroll
    for (int t = 0; t < 4; ++t) {
      v16bf b = frag_b_nt(&Bs[(t * 16) * LD], LD);
      acc[t] = wmma_bf16(a, b, acc[t]);
    }
    __syncthreads();
  }

  // epilogue: D layout -> lane n = lane&15, rows r + 8*(lane>=16)
#pragma unroll
  for (int t = 0; t < 4; ++t) {
    const int n  = n0 + t * 16 + (lane & 15);
    const float bn = bias[n];
    const int mb = m0 + wv * 16 + ((lane & 16) ? 8 : 0);
#pragma unroll
    for (int r = 0; r < 8; ++r) {
      const float v = acc[t][r] + bn;
      const size_t idx = (size_t)(mb + r) * N + n;
      if constexpr (OUT_BF16) ((bf16*)Op)[idx] = (bf16)v;
      else                    ((float*)Op)[idx] = v;
    }
  }
}

// ---------------------------------------------------------------------------
// Attention: one block per (b, h, w). 256 threads = 8 waves.
//   Qp[64,64] = SCALE * maxpool4(Q)  (interleaved groups s = g*64 + i)
//   S = Qp @ K^T  -> softmax rows -> attn (f32 to d_out) and P (bf16)
//   O = P @ V     -> bf16 to [b, q*32+w, h*64+hd]
// Dynamic LDS (147KB): K(36864) | V(36864) | Qp(9216) | S(67584); P overlays K.
// K/V rows staged with async global->LDS b128 (no conversion needed).
// ---------------------------------------------------------------------------
__global__ __launch_bounds__(256) void attn_kernel(
    const bf16* __restrict__ qkv, float* __restrict__ attn_out,
    bf16* __restrict__ oexp) {
  constexpr int LDK = 72;    // bf16 row stride for K/V/Qp (144B, 16B-aligned)
  constexpr int LDP = 264;   // f32 row stride for S; bf16 row stride for P

  extern __shared__ __align__(16) char smem[];
  bf16*  Ks = (bf16*)smem;                // [256][72]
  bf16*  Vs = (bf16*)(smem + 36864);      // [256][72]
  bf16*  Qp = (bf16*)(smem + 73728);      // [64][72]
  float* Sf = (float*)(smem + 82944);     // [64][264]
  bf16*  Ps = Ks;                         // overlay: [64][264] bf16 fits in K region

  const int tid  = threadIdx.x;
  const int wv   = tid >> 5;
  const int lane = tid & 31;
  const int w = blockIdx.x & 31;
  const int h = (blockIdx.x >> 5) % Hh;
  const int b = blockIdx.x / (NW * Hh);

  // ---- stage K and V rows: async global->LDS, 128B per row ----
  {
    const int s = tid;  // 0..255
    const size_t base = (size_t)(b * Tt + s * NW + w) * QKV_N + h * HD;
    const bf16* ksrc = qkv + base + Dd;
    const bf16* vsrc = qkv + base + 2 * Dd;
    const unsigned klo = lds_off32(&Ks[s * LDK]);
    const unsigned vlo = lds_off32(&Vs[s * LDK]);
    async_ld_b128<0>(klo, ksrc);   async_ld_b128<16>(klo, ksrc);
    async_ld_b128<32>(klo, ksrc);  async_ld_b128<48>(klo, ksrc);
    async_ld_b128<64>(klo, ksrc);  async_ld_b128<80>(klo, ksrc);
    async_ld_b128<96>(klo, ksrc);  async_ld_b128<112>(klo, ksrc);
    async_ld_b128<0>(vlo, vsrc);   async_ld_b128<16>(vlo, vsrc);
    async_ld_b128<32>(vlo, vsrc);  async_ld_b128<48>(vlo, vsrc);
    async_ld_b128<64>(vlo, vsrc);  async_ld_b128<80>(vlo, vsrc);
    async_ld_b128<96>(vlo, vsrc);  async_ld_b128<112>(vlo, vsrc);
  }

  // ---- Q max-pool over 4 interleaved groups, fold in SCALE ----
  {
    const int i = tid & 63;   // pooled row
    const int c = tid >> 6;   // 16-element chunk
    float m[16];
#pragma unroll
    for (int j = 0; j < 16; ++j) m[j] = -1e30f;
#pragma unroll
    for (int g = 0; g < 4; ++g) {
      const int s = g * 64 + i;
      const uint4* qsrc = (const uint4*)(
          qkv + (size_t)(b * Tt + s * NW + w) * QKV_N + h * HD + c * 16);
      union { uint4 v[2]; bf16 hx[16]; } u;
      u.v[0] = qsrc[0];
      u.v[1] = qsrc[1];
#pragma unroll
      for (int j = 0; j < 16; ++j) m[j] = fmaxf(m[j], (float)u.hx[j]);
    }
    bf16* qdst = &Qp[i * LDK + c * 16];
#pragma unroll
    for (int j = 0; j < 16; ++j) qdst[j] = (bf16)(m[j] * ATT_SCALE);
  }
  wait_async0();
  __syncthreads();

  // ---- S = Qp @ K^T  (64x256, K-dim 64): wave -> 4 row tiles x 2 col tiles ----
  {
    v8f sacc[8] = {};
#pragma unroll
    for (int kk = 0; kk < HD; kk += 32) {
#pragma unroll
      for (int rt = 0; rt < 4; ++rt) {
        v16bf a = frag_a(&Qp[(rt * 16) * LDK + kk], LDK);
#pragma unroll
        for (int cc = 0; cc < 2; ++cc) {
          v16bf bb = frag_b_nt(&Ks[((wv * 2 + cc) * 16) * LDK + kk], LDK);
          sacc[rt * 2 + cc] = wmma_bf16(a, bb, sacc[rt * 2 + cc]);
        }
      }
    }
#pragma unroll
    for (int rt = 0; rt < 4; ++rt)
#pragma unroll
      for (int cc = 0; cc < 2; ++cc) {
        const int n  = (wv * 2 + cc) * 16 + (lane & 15);
        const int mb = rt * 16 + ((lane & 16) ? 8 : 0);
#pragma unroll
        for (int r = 0; r < 8; ++r) Sf[(mb + r) * LDP + n] = sacc[rt * 2 + cc][r];
      }
  }
  __syncthreads();

  // ---- softmax: 4 lanes per row, shfl_xor row reductions ----
  {
    const int row = tid >> 2, q4 = tid & 3;
    float* srow = &Sf[row * LDP + q4 * 64];
    float rmax = -1e30f;
#pragma unroll
    for (int j = 0; j < 64; ++j) rmax = fmaxf(rmax, srow[j]);
    rmax = fmaxf(rmax, __shfl_xor(rmax, 1));
    rmax = fmaxf(rmax, __shfl_xor(rmax, 2));
    float rsum = 0.f;
#pragma unroll
    for (int j = 0; j < 64; ++j) {
      float e = __expf(srow[j] - rmax);
      srow[j] = e;
      rsum += e;
    }
    rsum += __shfl_xor(rsum, 1);
    rsum += __shfl_xor(rsum, 2);
    const float inv = 1.0f / rsum;
    float* gout =
        attn_out + (((size_t)((b * Hh + h) * NW + w) * QL + row) * SEQ + q4 * 64);
    bf16* prow = &Ps[row * LDP + q4 * 64];
#pragma unroll
    for (int j = 0; j < 64; ++j) {
      const float p = srow[j] * inv;
      gout[j] = p;              // attn output (f32), coalesced 256B chunks
      prow[j] = (bf16)p;        // P for the PV gemm
    }
  }
  __syncthreads();

  // ---- O = P @ V (64x64, K-dim 256): wave -> 1 row tile x 2 col tiles ----
  {
    const int rt = wv >> 1;
    const int nb = (wv & 1) * 2;
    v8f oacc[2] = {};
#pragma unroll
    for (int kk = 0; kk < SEQ; kk += 32) {
      v16bf a = frag_a(&Ps[(rt * 16) * LDP + kk], LDP);
#pragma unroll
      for (int cc = 0; cc < 2; ++cc) {
        v16bf bb = frag_b_nn(&Vs[kk * LDK + (nb + cc) * 16], LDK);
        oacc[cc] = wmma_bf16(a, bb, oacc[cc]);
      }
    }
#pragma unroll
    for (int cc = 0; cc < 2; ++cc) {
      const int n  = (nb + cc) * 16 + (lane & 15);
      const int mb = rt * 16 + ((lane & 16) ? 8 : 0);
#pragma unroll
      for (int r = 0; r < 8; ++r) {
        const int q = mb + r;
        const size_t idx = (size_t)(b * 2048 + q * NW + w) * Dd + h * HD + n;
        oexp[idx] = (bf16)oacc[cc][r];
      }
    }
  }
}

// ---------------------------------------------------------------------------

extern "C" void kernel_launch(void* const* d_in, const int* in_sizes, int n_in,
                              void* d_out, int out_size, void* d_ws, size_t ws_size,
                              hipStream_t stream) {
  (void)in_sizes; (void)n_in; (void)out_size; (void)ws_size;
  const float* emb    = (const float*)d_in[0];
  const float* qkv_w  = (const float*)d_in[1];
  const float* qkv_b  = (const float*)d_in[2];
  const float* proj_w = (const float*)d_in[3];
  const float* proj_b = (const float*)d_in[4];

  float* out  = (float*)d_out;                               // [8, 2048, 768]
  float* attn = out + (size_t)Bb * 2048 * Dd;                // [8,12,32,64,256]

  bf16* qkv  = (bf16*)d_ws;                                  // [8,8192,2304] bf16
  bf16* oexp = (bf16*)((char*)d_ws + (size_t)Bb * Tt * QKV_N * sizeof(bf16));

  // 1) QKV projection: [65536,768] x [2304,768]^T + bias -> bf16
  gemm_nt_bias<true, true>
      <<<dim3((Bb * Tt) / 128, QKV_N / 64), 256, 0, stream>>>(
          emb, qkv_w, qkv_b, qkv, Bb * Tt, QKV_N, Dd);

  // 2) windowed attention (3072 blocks), 147KB dynamic LDS
  attn_kernel<<<dim3(Bb * Hh * NW), 256, 150528, stream>>>(qkv, attn, oexp);

  // 3) output projection: [16384,768] x [768,768]^T + bias -> f32
  gemm_nt_bias<false, false>
      <<<dim3((Bb * 2048) / 128, Dd / 64), 256, 0, stream>>>(
          oexp, proj_w, proj_b, out, Bb * 2048, Dd, Dd);
}
